// SDM_25640954757319
// MI455X (gfx1250) — matrix operand order repaired and checked by reading, not compiled, not run on hardware
//
#include <hip/hip_runtime.h>
#include <hip/hip_bf16.h>
#include <math.h>

typedef unsigned short ushort_t;
typedef unsigned int   uint_t;
typedef __attribute__((ext_vector_type(16))) __bf16 v16bf;
typedef __attribute__((ext_vector_type(8)))  float  v8f;
typedef int v4i_t __attribute__((vector_size(16)));

#define NTOK 50176   // 16 * 56 * 56
#define LDT  40      // LDS tile row stride (ushorts): 80B, 16B aligned, conflict-friendly

// ---- async global->LDS path (CDNA5), guarded so the build never breaks ----
#if defined(__has_builtin)
#if __has_builtin(__builtin_amdgcn_global_load_async_to_lds_b128)
#define HAVE_ASYNC_LDS 1
#endif
#endif

#if HAVE_ASYNC_LDS
#define ASYNC_B128(gp, lp)                                                    \
    __builtin_amdgcn_global_load_async_to_lds_b128(                           \
        (__attribute__((address_space(1))) v4i_t*)(gp),                       \
        (__attribute__((address_space(3))) v4i_t*)(lp), 0, 0)
__device__ __forceinline__ void wait_async_all() {
#if __has_builtin(__builtin_amdgcn_s_wait_asynccnt)
    __builtin_amdgcn_s_wait_asynccnt(0);
#else
    asm volatile("s_wait_asynccnt 0" ::: "memory");
#endif
}
#else
__device__ __forceinline__ void wait_async_all() {}
#endif

__device__ __forceinline__ ushort_t f2bf(float f) {
    uint_t u = __float_as_uint(f);
    u += 0x7FFFu + ((u >> 16) & 1u);
    return (ushort_t)(u >> 16);
}

union ABFrag { v16bf v; uint_t u[8]; };

// A-fragment K offset for pair e (0..7), lane-half hi (0/1): per CDNA5 16-bit A layout
__device__ __forceinline__ int kA(int e, int hi) {
    return (e < 4 ? 2 * e : 16 + 2 * (e - 4)) + hi * 8;
}

// ---------------------------------------------------------------------------
// Elementwise / conversion kernels
// ---------------------------------------------------------------------------
__global__ void cvt_bf16_kernel(const float* __restrict__ in, ushort_t* __restrict__ out, int n) {
    int i = blockIdx.x * blockDim.x + threadIdx.x;
    if (i < n) out[i] = f2bf(in[i]);
}

// conv weight (Cout, Cin, 3, 3) f32 -> (Cout, kh, kw, Cin) bf16
__global__ void cvt_convw_kernel(const float* __restrict__ in, ushort_t* __restrict__ out, int nout) {
    int i = blockIdx.x * blockDim.x + threadIdx.x;
    int total = nout * 1728;
    if (i >= total) return;
    int n = i / 1728, rem = i % 1728;
    int kh = rem / 576, kw = (rem / 192) % 3, c = rem % 192;
    out[i] = f2bf(in[((size_t)(n * 192 + c) * 3 + kh) * 3 + kw]);
}

// x (B,C,H,W) f32 -> (token, C) bf16
__global__ void xt_kernel(const float* __restrict__ x, ushort_t* __restrict__ out) {
    int i = blockIdx.x * blockDim.x + threadIdx.x;
    if (i >= NTOK * 192) return;
    int m = i / 192, c = i % 192;
    int b = m / 3136, p = m % 3136;
    out[i] = f2bf(x[((size_t)b * 192 + c) * 3136 + p]);
}

// buf0 (B,H,W,C) f32 -> ds2 (B,C,H,W) f32 + bf16 NHWC copy for conv
__global__ void out2_kernel(const float* __restrict__ buf0, float* __restrict__ ds2,
                            ushort_t* __restrict__ xb) {
    int i = blockIdx.x * blockDim.x + threadIdx.x;
    if (i >= NTOK * 192) return;
    int m = i / 192, c = i % 192;
    float v = buf0[i];
    int b = m / 3136, p = m % 3136;
    ds2[((size_t)b * 192 + c) * 3136 + p] = v;
    xb[i] = f2bf(v);
}

// LayerNorm over C=192. mode 0: *g+b + pos -> f32 buf (patch embed)
// mode 1: *g+b, shifted-window partition -> bf16 (dest index = shifted raster token)
// mode 2: *g+b -> bf16 token order
__global__ void __launch_bounds__(64)
ln_kernel(const float* __restrict__ in, const float* __restrict__ g, const float* __restrict__ bta,
          const float* __restrict__ pos, float* __restrict__ outF, ushort_t* __restrict__ outB,
          int mode, int shift) {
    __shared__ float rs[64], rq[64];
    int m = blockIdx.x;
    int t = threadIdx.x;
    size_t srcTok = m;
    if (mode == 1 && shift) {
        int b = m / 3136, p = m % 3136;
        int h = p / 56, w = p % 56;
        int hs = h + shift; if (hs >= 56) hs -= 56;
        int ws2 = w + shift; if (ws2 >= 56) ws2 -= 56;
        srcTok = (size_t)b * 3136 + hs * 56 + ws2;
    }
    const float* xr = in + srcTok * 192;
    float v0 = xr[t], v1 = xr[t + 64], v2 = xr[t + 128];
    rs[t] = v0 + v1 + v2;
    rq[t] = v0 * v0 + v1 * v1 + v2 * v2;
    __syncthreads();
    for (int o = 32; o > 0; o >>= 1) {
        if (t < o) { rs[t] += rs[t + o]; rq[t] += rq[t + o]; }
        __syncthreads();
    }
    float mean = rs[0] * (1.f / 192.f);
    float var  = rq[0] * (1.f / 192.f) - mean * mean;
    float rstd = rsqrtf(var + 1e-5f);
    float o0 = (v0 - mean) * rstd * g[t]       + bta[t];
    float o1 = (v1 - mean) * rstd * g[t + 64]  + bta[t + 64];
    float o2 = (v2 - mean) * rstd * g[t + 128] + bta[t + 128];
    if (mode == 0) {
        const float* pp = pos + (size_t)(m % 3136) * 192;
        float* o = outF + (size_t)m * 192;
        o[t] = o0 + pp[t]; o[t + 64] = o1 + pp[t + 64]; o[t + 128] = o2 + pp[t + 128];
    } else if (mode == 2) {
        ushort_t* o = outB + (size_t)m * 192;
        o[t] = f2bf(o0); o[t + 64] = f2bf(o1); o[t + 128] = f2bf(o2);
    } else {
        int b = m / 3136, p = m % 3136, h = p / 56, w = p % 56;
        int wi = (h / 7) * 8 + (w / 7);
        int l  = (h % 7) * 7 + (w % 7);
        ushort_t* o = outB + (((size_t)b * 64 + wi) * 49 + l) * 192;
        o[t] = f2bf(o0); o[t + 64] = f2bf(o1); o[t + 128] = f2bf(o2);
    }
}

// ---------------------------------------------------------------------------
// bf16 WMMA GEMM: out = A(MxK) @ W(NxK)^T + bias
// 128x64 macro-tile, 4 waves, async double-buffered LDS pipeline.
// Each wave: 2 A-frags x 4 B-frags -> 8 WMMAs per 32-K step.
// ---------------------------------------------------------------------------
enum { EPI_F32B = 0, EPI_QKV = 1, EPI_GELU = 2, EPI_PROJ = 3, EPI_FC2 = 4 };

template <int EPI>
__global__ void __launch_bounds__(128)
gemm_bf16(const ushort_t* __restrict__ A, const ushort_t* __restrict__ Wm,
          const float* __restrict__ bias, int M, int N, int K,
          float* __restrict__ outF, ushort_t* __restrict__ outB,
          float* __restrict__ buf0, int shift) {
    __shared__ ushort_t As[2][128 * LDT];
    __shared__ ushort_t Bs[2][64 * LDT];
    const int t = threadIdx.x;
    const int m0 = blockIdx.y * 128, n0 = blockIdx.x * 64;
    const int wv = t >> 5, lane = t & 31, laneM = lane & 15, hi = lane >> 4;
    const int r = t >> 1, hh = t & 1;

    const ushort_t* Arow = A + (size_t)(m0 + t) * K;              // thread loads full A row
    const bool bval = (n0 + r) < N;
    const ushort_t* Brow = Wm + (size_t)(bval ? (n0 + r) : 0) * K + hh * 16;
    const uint4 z4 = make_uint4(0u, 0u, 0u, 0u);
    const int nk = K >> 5;

    // partial N-tile: keep those B rows zero in both buffers (never overwritten)
    if (!bval) {
        uint4* e0 = (uint4*)&Bs[0][r * LDT + hh * 16];
        uint4* e1 = (uint4*)&Bs[1][r * LDT + hh * 16];
        e0[0] = z4; e0[1] = z4; e1[0] = z4; e1[1] = z4;
    }

    auto issue_tile = [&](int kt, int pb) {
        ushort_t* la = &As[pb][t * LDT];
        const ushort_t* ga = Arow + kt * 32;
#if HAVE_ASYNC_LDS
#pragma unroll
        for (int c = 0; c < 4; ++c) ASYNC_B128(ga + c * 8, la + c * 8);
        if (bval) {
            ushort_t* lb = &Bs[pb][r * LDT + hh * 16];
            const ushort_t* gb = Brow + kt * 32;
            ASYNC_B128(gb, lb);
            ASYNC_B128(gb + 8, lb + 8);
        }
#else
        const uint4* s4 = (const uint4*)ga;
        uint4 a0 = s4[0], a1 = s4[1], a2 = s4[2], a3 = s4[3];
        uint4* d4 = (uint4*)la;
        d4[0] = a0; d4[1] = a1; d4[2] = a2; d4[3] = a3;
        if (bval) {
            const uint4* b4 = (const uint4*)(Brow + kt * 32);
            uint4 b0 = b4[0], b1 = b4[1];
            uint4* e4 = (uint4*)&Bs[pb][r * LDT + hh * 16];
            e4[0] = b0; e4[1] = b1;
        }
#endif
    };

    issue_tile(0, 0);
    int p = 0;
    v8f acc[2][4] = {};
    for (int kt = 0; kt < nk; ++kt) {
        wait_async_all();            // this wave's pending tile copies are in LDS
        __syncthreads();             // publish across all waves
        if (kt + 1 < nk) issue_tile(kt + 1, p ^ 1);   // copy engine runs under compute
        if (kt + 2 < nk) {           // global_prefetch_b8 of the tile after next
            __builtin_prefetch(Arow + (kt + 2) * 32, 0, 1);
            __builtin_prefetch(Brow + (kt + 2) * 32, 0, 1);
        }
        ABFrag a[2];
#pragma unroll
        for (int s = 0; s < 2; ++s)
#pragma unroll
            for (int e = 0; e < 8; ++e)
                a[s].u[e] = *(const uint_t*)&As[p][(wv * 32 + s * 16 + laneM) * LDT + kA(e, hi)];
#pragma unroll
        for (int j = 0; j < 4; ++j) {
            ABFrag b;
#pragma unroll
            for (int e = 0; e < 8; ++e)
                b.u[e] = *(const uint_t*)&Bs[p][(j * 16 + laneM) * LDT + kA(e, hi)];
#pragma unroll
            for (int s = 0; s < 2; ++s)
                acc[s][j] = __builtin_amdgcn_wmma_f32_16x16x32_bf16(false, a[s].v, false, b.v,
                                                                    (short)0, acc[s][j], false, false);
        }
        p ^= 1;
    }
#pragma unroll
    for (int j = 0; j < 4; ++j) {
        int n = n0 + j * 16 + laneM;
        if (n >= N) continue;
        float bv = bias[n];
#pragma unroll
        for (int s = 0; s < 2; ++s) {
#pragma unroll
            for (int rr = 0; rr < 8; ++rr) {
                int m = m0 + wv * 32 + s * 16 + rr + hi * 8;
                float val = acc[s][j][rr] + bv;
                if (EPI == EPI_F32B) {
                    outF[(size_t)m * N + n] = val;
                } else if (EPI == EPI_QKV) {
                    if (n < 192) val *= 0.2041241452319315f;  // HD^-0.5 folded into q
                    outB[(size_t)m * N + n] = f2bf(val);
                } else if (EPI == EPI_GELU) {
                    float gl = 0.5f * val * (1.f + erff(val * 0.7071067811865475f));
                    outB[(size_t)m * N + n] = f2bf(gl);
                } else if (EPI == EPI_FC2) {
                    buf0[(size_t)m * 192 + n] += val;
                } else {  // EPI_PROJ: window-reverse + unshift + residual add
                    int win = m / 49, l = m - win * 49;
                    int b = win >> 6, wi = win & 63;
                    int hp = (wi >> 3) * 7 + l / 7;
                    int wp = (wi & 7) * 7 + (l - (l / 7) * 7);
                    hp += shift; if (hp >= 56) hp -= 56;
                    wp += shift; if (wp >= 56) wp -= 56;
                    buf0[(((size_t)(b * 56 + hp)) * 56 + wp) * 192 + n] += val;
                }
            }
        }
    }
}

// ---------------------------------------------------------------------------
// Window attention: one WG per (window, head). L=49->64 pad, HD=24->32 pad.
// ---------------------------------------------------------------------------
__global__ void __launch_bounds__(128)
attn_kernel(const ushort_t* __restrict__ qkv, const float* __restrict__ tab,
            ushort_t* __restrict__ aout, int shift) {
    __shared__ ushort_t qs[64 * LDT];
    __shared__ ushort_t ks[64 * LDT];
    __shared__ ushort_t vs[32 * 72];   // [d][l]
    __shared__ ushort_t ps[64 * 72];   // [i][l] softmax probs bf16
    __shared__ float    sS[64 * 65];

    const int t = threadIdx.x;
    const int win = blockIdx.x >> 3, head = blockIdx.x & 7;
    const int wv = t >> 5, lane = t & 31, laneM = lane & 15, hi = lane >> 4;
    const size_t rowbase = (size_t)win * 49 * 576 + head * 24;

    for (int s = t; s < 64 * 32; s += 128) {
        int l = s >> 5, d = s & 31;
        ushort_t qv = 0, kv = 0;
        if (l < 49 && d < 24) {
            size_t o = rowbase + (size_t)l * 576 + d;
            qv = qkv[o];
            kv = qkv[o + 192];
        }
        qs[l * LDT + d] = qv;
        ks[l * LDT + d] = kv;
    }
    for (int s = t; s < 32 * 64; s += 128) {
        int d = s >> 6, l = s & 63;
        ushort_t vv = 0;
        if (l < 49 && d < 24) vv = qkv[rowbase + (size_t)l * 576 + 384 + d];
        vs[d * 72 + l] = vv;
    }
    __syncthreads();

    // S = q @ k^T  (wave's 16-row strip x 64 cols)
    v8f acc[4] = {};
    {
        ABFrag a;
#pragma unroll
        for (int e = 0; e < 8; ++e)
            a.u[e] = *(const uint_t*)&qs[(wv * 16 + laneM) * LDT + kA(e, hi)];
#pragma unroll
        for (int j = 0; j < 4; ++j) {
            ABFrag b;
#pragma unroll
            for (int e = 0; e < 8; ++e)
                b.u[e] = *(const uint_t*)&ks[(j * 16 + laneM) * LDT + kA(e, hi)];
            acc[j] = __builtin_amdgcn_wmma_f32_16x16x32_bf16(false, a.v, false, b.v,
                                                             (short)0, acc[j], false, false);
        }
    }
    // + relative position bias (+ shifted-window mask), stage to LDS
    const int wi = win & 63;
#pragma unroll
    for (int j = 0; j < 4; ++j) {
        int jc = j * 16 + laneM;
#pragma unroll
        for (int rr = 0; rr < 8; ++rr) {
            int i = wv * 16 + rr + hi * 8;
            float val = -1e30f;
            if (i < 49 && jc < 49) {
                int ih = i / 7, iw2 = i - ih * 7, jh = jc / 7, jw = jc - jh * 7;
                int ridx = (ih - jh + 6) * 13 + (iw2 - jw + 6);
                val = acc[j][rr] + tab[ridx * 8 + head];
                if (shift) {
                    int h_i = (wi >> 3) * 7 + ih, w_i = (wi & 7) * 7 + iw2;
                    int h_j = (wi >> 3) * 7 + jh, w_j = (wi & 7) * 7 + jw;
                    int thr = 56 - shift;
                    int ci = (h_i < 49 ? 0 : (h_i < thr ? 1 : 2)) * 3 + (w_i < 49 ? 0 : (w_i < thr ? 1 : 2));
                    int cj = (h_j < 49 ? 0 : (h_j < thr ? 1 : 2)) * 3 + (w_j < 49 ? 0 : (w_j < thr ? 1 : 2));
                    if (ci != cj) val -= 100.f;
                }
            }
            sS[i * 65 + jc] = val;
        }
    }
    __syncthreads();

    if (t < 64) {
        int i = t;
        if (i < 49) {
            float mx = -1e30f;
            for (int j = 0; j < 49; ++j) mx = fmaxf(mx, sS[i * 65 + j]);
            float sum = 0.f;
            for (int j = 0; j < 49; ++j) {
                float e = __expf(sS[i * 65 + j] - mx);
                sS[i * 65 + j] = e;
                sum += e;
            }
            float rsum = 1.f / sum;
            for (int j = 0; j < 49; ++j) ps[i * 72 + j] = f2bf(sS[i * 65 + j] * rsum);
            for (int j = 49; j < 64; ++j) ps[i * 72 + j] = 0;
        } else {
            for (int j = 0; j < 64; ++j) ps[i * 72 + j] = 0;
        }
    }
    __syncthreads();

    // O = P @ V : K=64 in 2 steps, N=32 (2 subtiles)
    v8f oacc[2] = {};
#pragma unroll
    for (int ks2 = 0; ks2 < 2; ++ks2) {
        ABFrag pa;
#pragma unroll
        for (int e = 0; e < 8; ++e)
            pa.u[e] = *(const uint_t*)&ps[(wv * 16 + laneM) * 72 + ks2 * 32 + kA(e, hi)];
#pragma unroll
        for (int j = 0; j < 2; ++j) {
            ABFrag vb;
#pragma unroll
            for (int e = 0; e < 8; ++e)
                vb.u[e] = *(const uint_t*)&vs[(j * 16 + laneM) * 72 + ks2 * 32 + kA(e, hi)];
            oacc[j] = __builtin_amdgcn_wmma_f32_16x16x32_bf16(false, pa.v, false, vb.v,
                                                              (short)0, oacc[j], false, false);
        }
    }
#pragma unroll
    for (int j = 0; j < 2; ++j) {
        int d = j * 16 + laneM;
        if (d >= 24) continue;
#pragma unroll
        for (int rr = 0; rr < 8; ++rr) {
            int l = wv * 16 + rr + hi * 8;
            if (l < 49)
                aout[((size_t)win * 49 + l) * 192 + head * 24 + d] = f2bf(oacc[j][rr]);
        }
    }
}

// ---------------------------------------------------------------------------
// 3x3 conv as implicit GEMM (K = 3*3*192 = 1728) + BN + ReLU
// 128x64 tile, async double-buffered. OUTMODE 0: bf16 NHWC (192ch); 1: f32 NCHW (96ch)
// ---------------------------------------------------------------------------
template <int OUTMODE>
__global__ void __launch_bounds__(128)
conv3_kernel(const ushort_t* __restrict__ inb, const ushort_t* __restrict__ Wm,
             const float* __restrict__ cb, const float* __restrict__ bg,
             const float* __restrict__ bb, const float* __restrict__ bm,
             const float* __restrict__ bv, int Nout,
             ushort_t* __restrict__ outB, float* __restrict__ outF) {
    __shared__ ushort_t As[2][128 * LDT];
    __shared__ ushort_t Bs[2][64 * LDT];
    const int t = threadIdx.x;
    const int m0 = blockIdx.y * 128, n0 = blockIdx.x * 64;
    const int wv = t >> 5, lane = t & 31, laneM = lane & 15, hi = lane >> 4;
    const int r = t >> 1, hh = t & 1;
    const int mtok = m0 + t;
    const int b = mtok / 3136, prem = mtok % 3136;
    const int oh = prem / 56, ow = prem % 56;
    const bool bval = (n0 + r) < Nout;
    const ushort_t* Brow = Wm + (size_t)(bval ? (n0 + r) : 0) * 1728 + hh * 16;
    const uint4 z4 = make_uint4(0u, 0u, 0u, 0u);
    const int nk = 54;  // 1728 / 32

    if (!bval) {
        uint4* e0 = (uint4*)&Bs[0][r * LDT + hh * 16];
        uint4* e1 = (uint4*)&Bs[1][r * LDT + hh * 16];
        e0[0] = z4; e0[1] = z4; e1[0] = z4; e1[1] = z4;
    }

    auto issue_tile = [&](int kt, int pb) {
        int kb = kt * 32;
        int kh = kb / 576, kw = (kb / 192) % 3, c0 = kb % 192;
        int ih = oh + kh - 1, iw = ow + kw - 1;
        ushort_t* la = &As[pb][t * LDT];
        bool valid = (ih >= 0 && ih < 56 && iw >= 0 && iw < 56);
        if (valid) {
            const ushort_t* ga = inb + (((size_t)(b * 56 + ih)) * 56 + iw) * 192 + c0;
#if HAVE_ASYNC_LDS
#pragma unroll
            for (int c = 0; c < 4; ++c) ASYNC_B128(ga + c * 8, la + c * 8);
#else
            const uint4* s4 = (const uint4*)ga;
            uint4 a0 = s4[0], a1 = s4[1], a2 = s4[2], a3 = s4[3];
            uint4* d4 = (uint4*)la;
            d4[0] = a0; d4[1] = a1; d4[2] = a2; d4[3] = a3;
#endif
        } else {  // halo: zero-fill with plain DS stores
            uint4* d4 = (uint4*)la;
            d4[0] = z4; d4[1] = z4; d4[2] = z4; d4[3] = z4;
        }
        if (bval) {
            const ushort_t* gb = Brow + kt * 32;
            ushort_t* lb = &Bs[pb][r * LDT + hh * 16];
#if HAVE_ASYNC_LDS
            ASYNC_B128(gb, lb);
            ASYNC_B128(gb + 8, lb + 8);
#else
            const uint4* b4 = (const uint4*)gb;
            uint4 b0 = b4[0], b1 = b4[1];
            uint4* e4 = (uint4*)lb;
            e4[0] = b0; e4[1] = b1;
#endif
        }
    };

    issue_tile(0, 0);
    int p = 0;
    v8f acc[2][4] = {};
    for (int kt = 0; kt < nk; ++kt) {
        wait_async_all();
        __syncthreads();
        if (kt + 1 < nk) issue_tile(kt + 1, p ^ 1);
        if (kt + 2 < nk) __builtin_prefetch(Brow + (kt + 2) * 32, 0, 1);
        ABFrag a[2];
#pragma unroll
        for (int s = 0; s < 2; ++s)
#pragma unroll
            for (int e = 0; e < 8; ++e)
                a[s].u[e] = *(const uint_t*)&As[p][(wv * 32 + s * 16 + laneM) * LDT + kA(e, hi)];
#pragma unroll
        for (int j = 0; j < 4; ++j) {
            ABFrag bfr;
#pragma unroll
            for (int e = 0; e < 8; ++e)
                bfr.u[e] = *(const uint_t*)&Bs[p][(j * 16 + laneM) * LDT + kA(e, hi)];
#pragma unroll
            for (int s = 0; s < 2; ++s)
                acc[s][j] = __builtin_amdgcn_wmma_f32_16x16x32_bf16(false, a[s].v, false, bfr.v,
                                                                    (short)0, acc[s][j], false, false);
        }
        p ^= 1;
    }
#pragma unroll
    for (int j = 0; j < 4; ++j) {
        int n = n0 + j * 16 + laneM;
        if (n >= Nout) continue;
        float sc = bg[n] * rsqrtf(bv[n] + 1e-5f);
        float sh = bb[n] - bm[n] * sc + cb[n] * sc;
#pragma unroll
        for (int s = 0; s < 2; ++s) {
#pragma unroll
            for (int rr = 0; rr < 8; ++rr) {
                int m = m0 + wv * 32 + s * 16 + rr + hi * 8;
                float val = fmaxf(acc[s][j][rr] * sc + sh, 0.f);
                if (OUTMODE == 0) {
                    outB[(size_t)m * 192 + n] = f2bf(val);
                } else {
                    int b2 = m / 3136, p2 = m % 3136;
                    outF[((size_t)b2 * 96 + n) * 3136 + p2] = val;
                }
            }
        }
    }
}

// ---------------------------------------------------------------------------
extern "C" void kernel_launch(void* const* d_in, const int* in_sizes, int n_in,
                              void* d_out, int out_size, void* d_ws, size_t ws_size,
                              hipStream_t stream) {
    (void)in_sizes; (void)n_in; (void)out_size; (void)ws_size;
    const float* x    = (const float*)d_in[0];
    const float* pos  = (const float*)d_in[1];
    const float* projw = (const float*)d_in[2];
    const float* projb = (const float*)d_in[3];
    const float* peg  = (const float*)d_in[4];
    const float* peb  = (const float*)d_in[5];
    const float* n1g  = (const float*)d_in[6];
    const float* n1b  = (const float*)d_in[7];
    const float* qkvw = (const float*)d_in[8];
    const float* qkvb = (const float*)d_in[9];
    const float* apw  = (const float*)d_in[10];
    const float* apb  = (const float*)d_in[11];
    const float* rtab = (const float*)d_in[12];
    const float* n2g  = (const float*)d_in[13];
    const float* n2b  = (const float*)d_in[14];
    const float* f1w  = (const float*)d_in[15];
    const float* f1b  = (const float*)d_in[16];
    const float* f2w  = (const float*)d_in[17];
    const float* f2b  = (const float*)d_in[18];
    const float* c1w  = (const float*)d_in[19];
    const float* c1b  = (const float*)d_in[20];
    const float* bn1g = (const float*)d_in[21];
    const float* bn1b = (const float*)d_in[22];
    const float* bn1m = (const float*)d_in[23];
    const float* bn1v = (const float*)d_in[24];
    const float* c2w  = (const float*)d_in[25];
    const float* c2b  = (const float*)d_in[26];
    const float* bn2g = (const float*)d_in[27];
    const float* bn2b = (const float*)d_in[28];
    const float* bn2m = (const float*)d_in[29];
    const float* bn2v = (const float*)d_in[30];

    // Workspace layout (time-aliased where lifetimes are disjoint)
    char* wsp = (char*)d_ws;
    float*    buf0 = (float*)wsp;      wsp += (size_t)NTOK * 192 * 4;   // residual stream f32
    ushort_t* xw   = (ushort_t*)wsp;   wsp += (size_t)NTOK * 192 * 2;   // LN1 windows / conv input bf16
    ushort_t* aout = (ushort_t*)wsp;   wsp += (size_t)NTOK * 192 * 2;   // attn out / conv1 out bf16
    ushort_t* ybuf = (ushort_t*)wsp;   wsp += (size_t)NTOK * 192 * 2;   // LN2 out bf16
    char*     big  = wsp;              wsp += (size_t)NTOK * 768 * 2;   // tbuf f32 | qkv bf16 | fc1 act bf16
    float*    tbuf   = (float*)big;
    ushort_t* qkvbuf = (ushort_t*)big;
    ushort_t* hbuf   = (ushort_t*)big;
    ushort_t* wpe  = (ushort_t*)wsp;   wsp += (size_t)36864 * 2;
    ushort_t* wqkv = (ushort_t*)wsp;   wsp += (size_t)442368 * 2;
    ushort_t* wproj = (ushort_t*)wsp;  wsp += (size_t)147456 * 2;
    ushort_t* wfc1 = (ushort_t*)wsp;   wsp += (size_t)589824 * 2;
    ushort_t* wfc2 = (ushort_t*)wsp;   wsp += (size_t)589824 * 2;
    ushort_t* wc1  = (ushort_t*)wsp;   wsp += (size_t)331776 * 2;
    ushort_t* wc2  = (ushort_t*)wsp;

    float* outConv = (float*)d_out;                       // (16, 96, 56, 56)
    float* outDs2  = outConv + (size_t)16 * 96 * 3136;    // (16, 192, 56, 56)

    // Weight conversions (f32 -> bf16, conv weights also reordered)
    auto gcd = [](int n) { return (n + 255) / 256; };
    cvt_bf16_kernel<<<gcd(36864), 256, 0, stream>>>(projw, wpe, 36864);
    cvt_bf16_kernel<<<gcd(442368), 256, 0, stream>>>(qkvw, wqkv, 442368);
    cvt_bf16_kernel<<<gcd(147456), 256, 0, stream>>>(apw, wproj, 147456);
    cvt_bf16_kernel<<<gcd(589824), 256, 0, stream>>>(f1w, wfc1, 589824);
    cvt_bf16_kernel<<<gcd(589824), 256, 0, stream>>>(f2w, wfc2, 589824);
    cvt_convw_kernel<<<gcd(331776), 256, 0, stream>>>(c1w, wc1, 192);
    cvt_convw_kernel<<<gcd(165888), 256, 0, stream>>>(c2w, wc2, 96);

    // Patch embed: x -> bf16 tokens, GEMM, LN + pos_embed -> buf0
    xt_kernel<<<gcd(NTOK * 192), 256, 0, stream>>>(x, xw);
    gemm_bf16<EPI_F32B><<<dim3(3, 392), 128, 0, stream>>>(xw, wpe, projb, NTOK, 192, 192,
                                                          tbuf, nullptr, nullptr, 0);
    ln_kernel<<<NTOK, 64, 0, stream>>>(tbuf, peg, peb, pos, buf0, nullptr, 0, 0);

    for (int blk = 0; blk < 4; ++blk) {
        int shift = (blk & 1) ? 3 : 0;
        ln_kernel<<<NTOK, 64, 0, stream>>>(buf0, n1g + blk * 192, n1b + blk * 192,
                                           nullptr, nullptr, xw, 1, shift);
        gemm_bf16<EPI_QKV><<<dim3(9, 392), 128, 0, stream>>>(
            xw, wqkv + (size_t)blk * 576 * 192, qkvb + blk * 576, NTOK, 576, 192,
            nullptr, qkvbuf, nullptr, 0);
        attn_kernel<<<8192, 128, 0, stream>>>(qkvbuf, rtab + (size_t)blk * 169 * 8, aout, shift);
        gemm_bf16<EPI_PROJ><<<dim3(3, 392), 128, 0, stream>>>(
            aout, wproj + (size_t)blk * 36864, apb + blk * 192, NTOK, 192, 192,
            nullptr, nullptr, buf0, shift);
        ln_kernel<<<NTOK, 64, 0, stream>>>(buf0, n2g + blk * 192, n2b + blk * 192,
                                           nullptr, nullptr, ybuf, 2, 0);
        gemm_bf16<EPI_GELU><<<dim3(12, 392), 128, 0, stream>>>(
            ybuf, wfc1 + (size_t)blk * 768 * 192, f1b + blk * 768, NTOK, 768, 192,
            nullptr, hbuf, nullptr, 0);
        gemm_bf16<EPI_FC2><<<dim3(3, 392), 128, 0, stream>>>(
            hbuf, wfc2 + (size_t)blk * 768 * 192, f2b + blk * 192, NTOK, 192, 768,
            nullptr, nullptr, buf0, 0);
    }

    // ds2 output + bf16 NHWC copy for conv path
    out2_kernel<<<gcd(NTOK * 192), 256, 0, stream>>>(buf0, outDs2, xw);
    conv3_kernel<0><<<dim3(3, 392), 128, 0, stream>>>(xw, wc1, c1b, bn1g, bn1b, bn1m, bn1v,
                                                      192, aout, nullptr);
    conv3_kernel<1><<<dim3(2, 392), 128, 0, stream>>>(aout, wc2, c2b, bn2g, bn2b, bn2m, bn2v,
                                                      96, nullptr, outConv);
}
